// net_81939386073095
// MI455X (gfx1250) — matrix operand order repaired and checked by reading, not compiled, not run on hardware
//
#include <hip/hip_runtime.h>
#include <stdint.h>

// ---------------------------------------------------------------------------
// CDNA5 / gfx1250 WMMA helpers (wave32).
// A: 16x32 f16, lane L: row = L&15; halves 0-7 -> K = 8*(L>>4)+[0..8),
//                                 halves 8-15 -> K = 16+8*(L>>4)+[0..8).
// B: 32x16 f16, lane L: col = L&15; same K mapping as A.
// C/D: f32 16x16, VGPR r: row = r + 8*(L>>4), col = L&15.
// ---------------------------------------------------------------------------
typedef __attribute__((ext_vector_type(16))) _Float16 v16h;
typedef __attribute__((ext_vector_type(8)))  float    v8f;

union V16 { v16h v; _Float16 h[16]; uint4 q[2]; };

__device__ __forceinline__ v8f wmma_f16(v16h a, v16h b, v8f c) {
  return __builtin_amdgcn_wmma_f32_16x16x32_f16(false, a, false, b,
                                                (short)0, c, false, false);
}

// One 16x16 output tile of (A[16xK] @ B), A f16 row-major (LDS or global,
// leading dim lda multiple of 8), B given as per-output-column contiguous
// K-vectors (row length bld halves, multiple of 32).
__device__ __forceinline__ v8f tile_mm(const _Float16* shA, int lda,
                                       const _Float16* Wcols, long bld,
                                       int K, v8f acc) {
  const int lane = threadIdx.x & 31;
  const int half = lane >> 4;
  const int mn   = lane & 15;
  const _Float16* arow = shA + mn * lda;
  const _Float16* bcol = Wcols + (long)mn * bld;
  for (int kb = 0; kb < K; kb += 32) {
    V16 A, B;
    A.q[0] = *(const uint4*)(arow + kb + 8 * half);
    A.q[1] = *(const uint4*)(arow + kb + 16 + 8 * half);
    const _Float16* bp = bcol + kb;
    B.q[0] = *(const uint4*)(bp + 8 * half);
    B.q[1] = *(const uint4*)(bp + 16 + 8 * half);
    acc = wmma_f16(A.v, B.v, acc);
  }
  return acc;
}

__device__ __forceinline__ float fast_sigmoid(float p) {
  // v_exp_f32 + v_rcp_f32 (avoid IEEE div_scale/fma refinement chains)
  return __builtin_amdgcn_rcpf(1.f + __expf(-p));
}

#define V8F_ZERO {0.f,0.f,0.f,0.f,0.f,0.f,0.f,0.f}

#define NSMP   1024
#define SINDY  47968          // 64 + 64 + 2080 + 45760
#define ZOFF   0
#define DZOFF  65536
#define DZBOFF 131072
#define XBOFF  196608
#define DXBOFF 262144

// ---------------------------------------------------------------------------
__global__ void k_cvt_f16(const float* __restrict__ s, _Float16* __restrict__ d,
                          int n) {
  int i = blockIdx.x * blockDim.x + threadIdx.x;
  if (i < n) d[i] = (_Float16)s[i];
}

// Packed monomial index table: theta[n, m] = zaug[i]*zaug[j]*zaug[k] with
// zaug[64] == 1.  idx[m] = i | (j<<8) | (k<<16).  Matches the reference's
// [ones, z, triu quad, i<=j<=k cubic] column ordering.
__global__ void k_build_idx(uint32_t* __restrict__ idx) {
  int tid = blockIdx.x * blockDim.x + threadIdx.x;
  if (tid >= 64 * 64 * 64) return;
  int i = (tid >> 12) & 63, j = (tid >> 6) & 63, k = tid & 63;
  if (i <= j && j <= k) {
    int off3 = 0;
    for (int a = 0; a < i; ++a) off3 += ((64 - a) * (65 - a)) >> 1;
    int off2 = (j - i) * 64 - ((j * (j - 1) - i * (i - 1)) >> 1);
    idx[2208 + off3 + off2 + (k - j)] =
        (uint32_t)i | ((uint32_t)j << 8) | ((uint32_t)k << 16);
  }
  if (tid < 64)       idx[tid] = 64u | (64u << 8) | (64u << 16);
  else if (tid < 128) idx[tid] = (uint32_t)(tid - 64) | (64u << 8) | (64u << 16);
  if (tid < 4096) {
    int qi = tid >> 6, qj = tid & 63;
    if (qi <= qj)
      idx[128 + qi * 64 - ((qi * (qi - 1)) >> 1) + (qj - qi)] =
          (uint32_t)qi | ((uint32_t)qj << 8) | (64u << 16);
  }
}

// Encoder forward, 1 wave per 16 samples: z = W2 s(W1 s(W0 x + b0) + b1) + b2.
// sigma' emitted TRANSPOSED in f16 (g[feature][sample]) so the gram kernels
// can consume it with contiguous-K WMMA loads.
__global__ __launch_bounds__(32) void k_enc_fwd(
    const float* __restrict__ x,
    const _Float16* __restrict__ W0h, const float* __restrict__ b0,
    const _Float16* __restrict__ W1h, const float* __restrict__ b1,
    const _Float16* __restrict__ W2h, const float* __restrict__ b2,
    _Float16* __restrict__ g1t, _Float16* __restrict__ g2t,
    float* __restrict__ z_out, _Float16* __restrict__ zh) {
  __shared__ _Float16 shX[16 * 64];
  __shared__ _Float16 shH1[16 * 256];
  __shared__ _Float16 shH2[16 * 128];
  const int lane = threadIdx.x;
  const int s0 = blockIdx.x * 16;
  for (int t = lane; t < 16 * 64; t += 32) {
    int r = t >> 6, c = t & 63;
    shX[t] = (_Float16)x[(size_t)(s0 + r) * 64 + c];
  }
  __syncthreads();
  const int half = lane >> 4, n = lane & 15;
  for (int ct = 0; ct < 16; ++ct) {          // 64 -> 256, sigmoid
    v8f acc = V8F_ZERO;
    acc = tile_mm(shX, 64, W0h + (long)ct * 16 * 64, 64, 64, acc);
    int col = ct * 16 + n; float bias = b0[col];
#pragma unroll
    for (int r = 0; r < 8; ++r) {
      int m = r + 8 * half;
      float h = fast_sigmoid(acc[r] + bias);
      shH1[m * 256 + col] = (_Float16)h;
      g1t[(size_t)col * NSMP + s0 + m] = (_Float16)(h * (1.f - h));
    }
  }
  __syncthreads();
  for (int ct = 0; ct < 8; ++ct) {           // 256 -> 128, sigmoid
    v8f acc = V8F_ZERO;
    acc = tile_mm(shH1, 256, W1h + (long)ct * 16 * 256, 256, 256, acc);
    int col = ct * 16 + n; float bias = b1[col];
#pragma unroll
    for (int r = 0; r < 8; ++r) {
      int m = r + 8 * half;
      float h = fast_sigmoid(acc[r] + bias);
      shH2[m * 128 + col] = (_Float16)h;
      g2t[(size_t)col * NSMP + s0 + m] = (_Float16)(h * (1.f - h));
    }
  }
  __syncthreads();
  for (int ct = 0; ct < 4; ++ct) {           // 128 -> 64, linear
    v8f acc = V8F_ZERO;
    acc = tile_mm(shH2, 128, W2h + (long)ct * 16 * 128, 128, 128, acc);
    int col = ct * 16 + n; float bias = b2[col];
#pragma unroll
    for (int r = 0; r < 8; ++r) {
      int m = r + 8 * half;
      float zv = acc[r] + bias;
      z_out[(size_t)(s0 + m) * 64 + col] = zv;
      zh[(size_t)(s0 + m) * 64 + col] = (_Float16)zv;
    }
  }
}

// Decoder forward (f16 z in): xb = Wd2 s(Wd1 s(Wd0 z + b0) + b1) + b2.
__global__ __launch_bounds__(32) void k_dec_fwd(
    const _Float16* __restrict__ zh,
    const _Float16* __restrict__ W0h, const float* __restrict__ b0,
    const _Float16* __restrict__ W1h, const float* __restrict__ b1,
    const _Float16* __restrict__ W2h, const float* __restrict__ b2,
    _Float16* __restrict__ gd1t, _Float16* __restrict__ gd2t,
    float* __restrict__ xb) {
  __shared__ _Float16 shZ[16 * 64];
  __shared__ _Float16 shD1[16 * 128];
  __shared__ _Float16 shD2[16 * 256];
  const int lane = threadIdx.x;
  const int s0 = blockIdx.x * 16;
  for (int t = lane; t < 16 * 64; t += 32) {
    int r = t >> 6, c = t & 63;
    shZ[t] = zh[(size_t)(s0 + r) * 64 + c];
  }
  __syncthreads();
  const int half = lane >> 4, n = lane & 15;
  for (int ct = 0; ct < 8; ++ct) {           // 64 -> 128, sigmoid
    v8f acc = V8F_ZERO;
    acc = tile_mm(shZ, 64, W0h + (long)ct * 16 * 64, 64, 64, acc);
    int col = ct * 16 + n; float bias = b0[col];
#pragma unroll
    for (int r = 0; r < 8; ++r) {
      int m = r + 8 * half;
      float h = fast_sigmoid(acc[r] + bias);
      shD1[m * 128 + col] = (_Float16)h;
      gd1t[(size_t)col * NSMP + s0 + m] = (_Float16)(h * (1.f - h));
    }
  }
  __syncthreads();
  for (int ct = 0; ct < 16; ++ct) {          // 128 -> 256, sigmoid
    v8f acc = V8F_ZERO;
    acc = tile_mm(shD1, 128, W1h + (long)ct * 16 * 128, 128, 128, acc);
    int col = ct * 16 + n; float bias = b1[col];
#pragma unroll
    for (int r = 0; r < 8; ++r) {
      int m = r + 8 * half;
      float h = fast_sigmoid(acc[r] + bias);
      shD2[m * 256 + col] = (_Float16)h;
      gd2t[(size_t)col * NSMP + s0 + m] = (_Float16)(h * (1.f - h));
    }
  }
  __syncthreads();
  for (int ct = 0; ct < 4; ++ct) {           // 256 -> 64, linear
    v8f acc = V8F_ZERO;
    acc = tile_mm(shD2, 256, W2h + (long)ct * 16 * 256, 256, 256, acc);
    int col = ct * 16 + n; float bias = b2[col];
#pragma unroll
    for (int r = 0; r < 8; ++r) {
      int m = r + 8 * half;
      xb[(size_t)(s0 + m) * 64 + col] = acc[r] + bias;
    }
  }
}

// WMMA gram: G[p,q] = (1/N) sum_n At[p][n] * Bt[q][n].  Both operands are
// [feature][sample] f16 so K runs contiguous for A rows and B columns alike.
// One wave per 16x16 tile of G; K = 1024.
__global__ __launch_bounds__(32) void k_gram16(const _Float16* __restrict__ At,
                                               const _Float16* __restrict__ Bt,
                                               int Q, float* __restrict__ G) {
  const int p0 = blockIdx.y * 16, q0 = blockIdx.x * 16;
  v8f acc = V8F_ZERO;
  acc = tile_mm(At + (long)p0 * NSMP, NSMP, Bt + (long)q0 * NSMP, NSMP, NSMP,
                acc);
  const int lane = threadIdx.x, half = lane >> 4, n = lane & 15;
#pragma unroll
  for (int r = 0; r < 8; ++r) {
    int m = r + 8 * half;
    G[(size_t)(p0 + m) * Q + q0 + n] = acc[r] * (1.f / 1024.f);
  }
}

// T[a,q] = sum_p W2[a,p] * W1[p,q] * G[p,q]   (a in 0..63)
__global__ void k_jacT(const float* __restrict__ W2, int P,
                       const float* __restrict__ W1,
                       const float* __restrict__ G, int Q,
                       float* __restrict__ T) {
  int t = blockIdx.x * blockDim.x + threadIdx.x;
  if (t >= 64 * Q) return;
  int a = t / Q, q = t - a * Q;
  float s = 0.f;
  for (int p = 0; p < P; ++p) s += W2[a * P + p] * W1[p * Q + q] * G[p * Q + q];
  T[t] = s;
}

// Jt[b,a] = sum_q T[a,q] * W0[q,b]  (stored transposed in f16 for WMMA B use)
__global__ void k_jacJ(const float* __restrict__ T, int Q,
                       const float* __restrict__ W0,
                       _Float16* __restrict__ Jt) {
  int t = blockIdx.x * blockDim.x + threadIdx.x;
  if (t >= 64 * 64) return;
  int a = t >> 6, b = t & 63;
  float s = 0.f;
  for (int q = 0; q < Q; ++q) s += T[a * Q + q] * W0[q * 64 + b];
  Jt[b * 64 + a] = (_Float16)s;
}

// out[n,b] = sum_a in[n,a] * J[a,b]  with Jt[b][a] contiguous per column.
__global__ __launch_bounds__(32) void k_applyJ(const float* __restrict__ in,
                                               const _Float16* __restrict__ Jt,
                                               float* __restrict__ out) {
  __shared__ _Float16 shA[16 * 64];
  const int lane = threadIdx.x;
  const int s0 = blockIdx.x * 16;
  for (int t = lane; t < 16 * 64; t += 32) {
    int r = t >> 6, c = t & 63;
    shA[t] = (_Float16)in[(size_t)(s0 + r) * 64 + c];
  }
  __syncthreads();
  const int half = lane >> 4, n = lane & 15;
  for (int ct = 0; ct < 4; ++ct) {
    v8f acc = V8F_ZERO;
    acc = tile_mm(shA, 64, Jt + (long)ct * 16 * 64, 64, 64, acc);
    int col = ct * 16 + n;
#pragma unroll
    for (int r = 0; r < 8; ++r) {
      int m = r + 8 * half;
      out[(size_t)(s0 + m) * 64 + col] = acc[r];
    }
  }
}

// Fused SINDy GEMM: dzb = theta(z) @ E_W^T + E_b.  theta is generated on the
// fly into WMMA A-layout from the packed (i,j,k) table; the E_W panel for the
// current K-chunk (64 cols x 32 K = 4KB) is double-buffered into LDS with
// gfx1250 async loads (ASYNCcnt) and overlapped with monomial generation.
__global__ __launch_bounds__(32) void k_sindy(const float* __restrict__ z,
                                              const uint32_t* __restrict__ idx,
                                              const _Float16* __restrict__ EWh,
                                              const float* __restrict__ Eb,
                                              float* __restrict__ dzb) {
  __shared__ float zaug[16][66];
  __shared__ _Float16 Bbuf[2][64 * 32];     // [buf][col*32 + k]
  const int lane = threadIdx.x;
  const int s0 = blockIdx.x * 16;
  for (int t = lane; t < 16 * 64; t += 32) {
    int r = t >> 6, c = t & 63;
    zaug[r][c] = z[(size_t)(s0 + r) * 64 + c];
  }
  if (lane < 16) zaug[lane][64] = 1.f;
  __syncthreads();

  // 8 x 32 lanes x 16B async copies bring one 4KB E_W panel into LDS.
  auto prefetch = [&](int kb, int b) {
#pragma unroll
    for (int inst = 0; inst < 8; ++inst) {
      int it = inst * 32 + lane;
      int col = it >> 2, part = it & 3;               // 4 x 16B per column
      const _Float16* src = EWh + (size_t)col * SINDY + kb + part * 8;
      uint32_t dst = (uint32_t)(uintptr_t)(&Bbuf[b][col * 32 + part * 8]);
      asm volatile("global_load_async_to_lds_b128 %0, %1, off"
                   :: "v"(dst), "v"(src) : "memory");
    }
  };

  const int half = lane >> 4, n = lane & 15;
  const float* zr = zaug[n];                 // A row for this lane
  v8f acc[4] = {V8F_ZERO, V8F_ZERO, V8F_ZERO, V8F_ZERO};
  prefetch(0, 0);
  int buf = 0;
  for (int kb = 0; kb < SINDY; kb += 32) {
    if (kb + 32 < SINDY) {
      prefetch(kb + 32, buf ^ 1);
      // async loads complete in order: <=8 outstanding => previous panel done
      asm volatile("s_wait_asynccnt 0x8" ::: "memory");
    } else {
      asm volatile("s_wait_asynccnt 0x0" ::: "memory");
    }
    V16 A;
    const uint32_t* i0 = idx + kb + 8 * half;
    const uint32_t* i1 = i0 + 16;
#pragma unroll
    for (int t = 0; t < 8; ++t) {
      uint32_t p = i0[t];
      A.h[t] = (_Float16)(zr[p & 255] * zr[(p >> 8) & 255] * zr[(p >> 16) & 255]);
      uint32_t p2 = i1[t];
      A.h[8 + t] =
          (_Float16)(zr[p2 & 255] * zr[(p2 >> 8) & 255] * zr[(p2 >> 16) & 255]);
    }
#pragma unroll
    for (int ct = 0; ct < 4; ++ct) {
      const _Float16* bp = &Bbuf[buf][(ct * 16 + n) * 32];
      V16 B;
      B.q[0] = *(const uint4*)(bp + 8 * half);
      B.q[1] = *(const uint4*)(bp + 16 + 8 * half);
      acc[ct] = wmma_f16(A.v, B.v, acc[ct]);
    }
    buf ^= 1;
  }
#pragma unroll
  for (int ct = 0; ct < 4; ++ct) {
    int col = ct * 16 + n; float bias = Eb[col];
#pragma unroll
    for (int r = 0; r < 8; ++r) {
      int m = r + 8 * half;
      dzb[(size_t)(s0 + m) * 64 + col] = acc[ct][r] + bias;
    }
  }
}

// ---------------------------------------------------------------------------
extern "C" void kernel_launch(void* const* d_in, const int* in_sizes, int n_in,
                              void* d_out, int out_size, void* d_ws,
                              size_t ws_size, hipStream_t stream) {
  (void)in_sizes; (void)n_in; (void)out_size; (void)ws_size;
  const float* x    = (const float*)d_in[0];
  const float* dx   = (const float*)d_in[1];
  // d_in[2] = ddx : unused by the reference computation.
  const float* weW0 = (const float*)d_in[3];
  const float* weB0 = (const float*)d_in[4];
  const float* weW1 = (const float*)d_in[5];
  const float* weB1 = (const float*)d_in[6];
  const float* weW2 = (const float*)d_in[7];
  const float* weB2 = (const float*)d_in[8];
  const float* wdW0 = (const float*)d_in[9];
  const float* wdB0 = (const float*)d_in[10];
  const float* wdW1 = (const float*)d_in[11];
  const float* wdB1 = (const float*)d_in[12];
  const float* wdW2 = (const float*)d_in[13];
  const float* wdB2 = (const float*)d_in[14];
  const float* EW   = (const float*)d_in[15];
  const float* EB   = (const float*)d_in[16];
  float* out = (float*)d_out;

  // ~9 MB of workspace, carved with 256-B alignment.
  char* p = (char*)d_ws;
  auto carve = [&](size_t bytes) -> char* {
    char* r = p; p += (bytes + 255) & ~(size_t)255; return r;
  };
  float* Ge   = (float*)carve((size_t)128 * 256 * 4);
  float* Gd   = (float*)carve((size_t)256 * 128 * 4);
  float* Tb   = (float*)carve((size_t)64 * 256 * 4);
  _Float16* g1t  = (_Float16*)carve((size_t)256 * NSMP * 2);
  _Float16* g2t  = (_Float16*)carve((size_t)128 * NSMP * 2);
  _Float16* gd1t = (_Float16*)carve((size_t)128 * NSMP * 2);
  _Float16* gd2t = (_Float16*)carve((size_t)256 * NSMP * 2);
  _Float16* W0h  = (_Float16*)carve((size_t)256 * 64 * 2);
  _Float16* W1h  = (_Float16*)carve((size_t)128 * 256 * 2);
  _Float16* W2h  = (_Float16*)carve((size_t)64 * 128 * 2);
  _Float16* Wd0h = (_Float16*)carve((size_t)128 * 64 * 2);
  _Float16* Wd1h = (_Float16*)carve((size_t)256 * 128 * 2);
  _Float16* Wd2h = (_Float16*)carve((size_t)64 * 256 * 2);
  _Float16* EWh  = (_Float16*)carve((size_t)64 * SINDY * 2);
  _Float16* zh   = (_Float16*)carve((size_t)1024 * 64 * 2);
  _Float16* Jte  = (_Float16*)carve((size_t)64 * 64 * 2);
  _Float16* Jtd  = (_Float16*)carve((size_t)64 * 64 * 2);
  uint32_t* idx  = (uint32_t*)carve((size_t)SINDY * 4);

  auto cvt = [&](const float* s, _Float16* d, int n) {
    k_cvt_f16<<<(n + 255) / 256, 256, 0, stream>>>(s, d, n);
  };
  cvt(weW0, W0h, 256 * 64);
  cvt(weW1, W1h, 128 * 256);
  cvt(weW2, W2h, 64 * 128);
  cvt(wdW0, Wd0h, 128 * 64);
  cvt(wdW1, Wd1h, 256 * 128);
  cvt(wdW2, Wd2h, 64 * 256);
  cvt(EW, EWh, 64 * SINDY);
  k_build_idx<<<(64 * 64 * 64) / 256, 256, 0, stream>>>(idx);

  // Encoder path
  k_enc_fwd<<<64, 32, 0, stream>>>(x, W0h, weB0, W1h, weB1, W2h, weB2,
                                   g1t, g2t, out + ZOFF, zh);
  k_gram16<<<dim3(16, 8), 32, 0, stream>>>(g2t, g1t, 256, Ge);
  k_jacT<<<(64 * 256 + 255) / 256, 256, 0, stream>>>(weW2, 128, weW1, Ge, 256, Tb);
  k_jacJ<<<16, 256, 0, stream>>>(Tb, 256, weW0, Jte);
  k_applyJ<<<64, 32, 0, stream>>>(dx, Jte, out + DZOFF);

  // Decoder path
  k_dec_fwd<<<64, 32, 0, stream>>>(zh, Wd0h, wdB0, Wd1h, wdB1, Wd2h, wdB2,
                                   gd1t, gd2t, out + XBOFF);
  k_gram16<<<dim3(8, 16), 32, 0, stream>>>(gd2t, gd1t, 128, Gd);
  k_jacT<<<(64 * 128 + 255) / 256, 256, 0, stream>>>(wdW2, 256, wdW1, Gd, 128, Tb);
  k_jacJ<<<16, 256, 0, stream>>>(Tb, 128, wdW0, Jtd);

  // SINDy regression + dxb
  k_sindy<<<64, 32, 0, stream>>>(out + ZOFF, idx, EWh, EB, out + DZBOFF);
  k_applyJ<<<64, 32, 0, stream>>>(out + DZBOFF, Jtd, out + DXBOFF);
}
